// Sam3ViTLayer_73100343378044
// MI455X (gfx1250) — matrix operand order repaired and checked by reading, not compiled, not run on hardware
//
#include <hip/hip_runtime.h>
#include <hip/hip_bf16.h>
#include <math.h>

typedef __bf16 bf16_t;
typedef __attribute__((ext_vector_type(16))) __bf16 v16bf;
typedef __attribute__((ext_vector_type(8)))  __bf16 v8bf;
typedef __attribute__((ext_vector_type(8)))  float  v8f;

#define HS   1024
#define NH   16
#define HD   64
#define WSZ  24
#define S_WIN 576
#define NTOK 10368          // 2*72*72
#define FFN  4096

// ---------------------------------------------------------------------------
// WMMA helpers (CDNA5 wave32, V_WMMA_F32_16X16X32_BF16)
// ---------------------------------------------------------------------------
__device__ __forceinline__ v8f wmma_bf16(v16bf a, v16bf b, v8f c) {
  return __builtin_amdgcn_wmma_f32_16x16x32_bf16(false, a, false, b, (short)0, c,
                                                 false, false);
}

// A-matrix 16x32 bf16 fragment, contiguous k order (B fragments use the same):
// lane l: m = l%16 ; base = (l/16)*16 ; elements e: k = base+e
// 'row' must point at A[m][k_chunk_start]
__device__ __forceinline__ v16bf load_a_frag(const bf16_t* row, int lane) {
  int base = (lane >> 4) * 16;
  v8bf lo = *(const v8bf*)(row + base);
  v8bf hi = *(const v8bf*)(row + base + 8);
  return __builtin_shufflevector(lo, hi, 0, 1, 2, 3, 4, 5, 6, 7,
                                         8, 9, 10, 11, 12, 13, 14, 15);
}

// ---------------------------------------------------------------------------
// f32 -> bf16 weight conversion
// ---------------------------------------------------------------------------
__global__ __launch_bounds__(256) void cvt_bf16_kernel(const float* __restrict__ s,
                                                       bf16_t* __restrict__ d, int n) {
  int i = blockIdx.x * 256 + threadIdx.x;
  if (i < n) d[i] = (bf16_t)s[i];
}

// ---------------------------------------------------------------------------
// LayerNorm (one block per token).  PARTITION=true writes window-ordered rows.
// ---------------------------------------------------------------------------
template <bool PARTITION>
__global__ __launch_bounds__(256)
void ln_kernel(const float* __restrict__ x, const float* __restrict__ g,
               const float* __restrict__ b, bf16_t* __restrict__ out) {
  __shared__ float s_sum[256], s_sq[256];
  const int t = blockIdx.x;
  const float* row = x + (size_t)t * HS;
  float sum = 0.f, sq = 0.f;
  for (int i = threadIdx.x; i < HS; i += 256) {
    float v = row[i];
    sum += v; sq += v * v;
  }
  s_sum[threadIdx.x] = sum; s_sq[threadIdx.x] = sq;
  __syncthreads();
  for (int o = 128; o > 0; o >>= 1) {
    if (threadIdx.x < o) {
      s_sum[threadIdx.x] += s_sum[threadIdx.x + o];
      s_sq[threadIdx.x]  += s_sq[threadIdx.x + o];
    }
    __syncthreads();
  }
  float mean = s_sum[0] * (1.f / HS);
  float var  = s_sq[0] * (1.f / HS) - mean * mean;
  float rstd = rsqrtf(var + 1e-6f);

  size_t drow;
  if (PARTITION) {
    int bb = t / (72 * 72), rem = t - bb * 72 * 72;
    int y = rem / 72, xq = rem - y * 72;
    int wrow = ((bb * 9 + (y / WSZ) * 3 + (xq / WSZ)) * S_WIN) +
               (y % WSZ) * WSZ + (xq % WSZ);
    drow = (size_t)wrow * HS;
  } else {
    drow = (size_t)t * HS;
  }
  for (int i = threadIdx.x; i < HS; i += 256) {
    float v = (row[i] - mean) * rstd * g[i] + b[i];
    out[drow + i] = (bf16_t)v;
  }
}

// ---------------------------------------------------------------------------
// In-place 2D RoPE on window-ordered Q or K (bf16). One thread per (row,head,pair)
// ---------------------------------------------------------------------------
__global__ __launch_bounds__(256) void rope_kernel(bf16_t* __restrict__ qk) {
  int idx = blockIdx.x * 256 + threadIdx.x;
  if (idx >= NTOK * NH * 32) return;
  int row = idx / (NH * 32);
  int rem = idx - row * (NH * 32);
  int h = rem >> 5, p = rem & 31;
  int s = row % S_WIN;
  float pos = (p < 16) ? (float)(s % WSZ) : (float)(s / WSZ);
  pos *= (1.0f / 3.0f);                               // scale = GWS/WS = 8/24
  int j = (p < 16) ? p : p - 16;
  float freq = __powf(10000.0f, -(float)j * (1.0f / 16.0f));
  float ang = pos * freq;
  float sn, cs;
  __sincosf(ang, &sn, &cs);
  size_t o = (size_t)row * HS + h * HD + p * 2;
  float x0 = (float)qk[o], x1 = (float)qk[o + 1];
  qk[o]     = (bf16_t)(x0 * cs - x1 * sn);
  qk[o + 1] = (bf16_t)(x1 * cs + x0 * sn);
}

// ---------------------------------------------------------------------------
// Tiled WMMA GEMM: C[M,N] = A[M,K](bf16) x B[K,N](bf16) + bias, epilogue by MODE
//   MODE 0: store bf16
//   MODE 1: exact GELU, store bf16
//   MODE 2: window-unpartition row, add residual (original order), store f32
//   MODE 3: add residual (same order), store f32
// Block tile 128x128, 256 threads (8 waves: 4 M-groups x 2 N-groups).
// k-slab of 64 per barrier phase (two 32-wide WMMA chunks) + prefetch of next
// slab via global_prefetch_b8.
// ---------------------------------------------------------------------------
template <int MODE>
__global__ __launch_bounds__(256)
void gemm_kernel(const bf16_t* __restrict__ A, const bf16_t* __restrict__ B,
                 const float* __restrict__ bias, void* __restrict__ outp,
                 const float* __restrict__ res, int M, int N, int K) {
  __shared__ alignas(64) bf16_t Alds[2][128][32];   // [k-chunk][m][k] row-major
  __shared__ alignas(64) bf16_t Blds[2][128][32];   // [k-chunk][n][k] transposed

  const int tid = threadIdx.x;
  const int lane = tid & 31;
  const int wave = tid >> 5;
  const int wm = wave & 3;                          // 4 row groups of 32
  const int wn = wave >> 2;                         // 2 col groups of 64
  const int bm = blockIdx.y * 128;
  const int bn = blockIdx.x * 128;
  const int lm = lane & 15;
  const int lg = lane >> 4;

  v8f acc[2][4];
#pragma unroll
  for (int i = 0; i < 2; i++)
#pragma unroll
    for (int j = 0; j < 4; j++) acc[i][j] = v8f{0, 0, 0, 0, 0, 0, 0, 0};

  const int ar = tid >> 1, ah = (tid & 1) * 16;     // A-tile loader coords
  const int bk = tid >> 3, bj = (tid & 7) * 16;     // B-tile loader coords

  for (int k0 = 0; k0 < K; k0 += 64) {
    __syncthreads();
#pragma unroll
    for (int c = 0; c < 2; c++) {
      // A slab: 128 x 64, contiguous 32B per thread per chunk
      *(v16bf*)&Alds[c][ar][ah] =
          *(const v16bf*)(A + (size_t)(bm + ar) * K + k0 + c * 32 + ah);
      // B slab: read row-major, store transposed for contiguous B-fragments
      v16bf tmp = *(const v16bf*)(B + (size_t)(k0 + c * 32 + bk) * N + bn + bj);
#pragma unroll
      for (int e = 0; e < 16; e++) Blds[c][bj + e][bk] = tmp[e];
    }
    // software prefetch of the next k-slab (global_prefetch_b8)
    if (k0 + 64 < K) {
      __builtin_prefetch(A + (size_t)(bm + ar) * K + k0 + 64 + ah, 0, 1);
      __builtin_prefetch(B + (size_t)(k0 + 64 + bk) * N + bn + bj, 0, 1);
    }
    __syncthreads();

#pragma unroll
    for (int c = 0; c < 2; c++) {
      v16bf afrag[2];
#pragma unroll
      for (int mi = 0; mi < 2; mi++)
        afrag[mi] = load_a_frag(&Alds[c][wm * 32 + mi * 16 + lm][0], lane);
#pragma unroll
      for (int ni = 0; ni < 4; ni++) {
        v16bf bfrag = *(const v16bf*)&Blds[c][wn * 64 + ni * 16 + lm][lg * 16];
#pragma unroll
        for (int mi = 0; mi < 2; mi++)
          acc[mi][ni] = wmma_bf16(afrag[mi], bfrag, acc[mi][ni]);
      }
    }
  }

  // Epilogue (C/D layout: lane holds rows r + 8*lg, col = lm within 16x16 tile)
#pragma unroll
  for (int mi = 0; mi < 2; mi++) {
#pragma unroll
    for (int ni = 0; ni < 4; ni++) {
      int gn = bn + wn * 64 + ni * 16 + lm;
      float bv = bias[gn];
#pragma unroll
      for (int r = 0; r < 8; r++) {
        int gm = bm + wm * 32 + mi * 16 + r + lg * 8;
        float v = acc[mi][ni][r] + bv;
        if (MODE == 0) {
          ((bf16_t*)outp)[(size_t)gm * N + gn] = (bf16_t)v;
        } else if (MODE == 1) {
          float gl = 0.5f * v * (1.0f + erff(v * 0.70710678118654752f));
          ((bf16_t*)outp)[(size_t)gm * N + gn] = (bf16_t)gl;
        } else if (MODE == 2) {
          int w = gm / S_WIN, s = gm - w * S_WIN;
          int b = w / 9, wr = w - b * 9;
          int wy = wr / 3, wx = wr - wy * 3;
          int ly = s / WSZ, lx = s - ly * WSZ;
          int orig = (b * 72 + wy * WSZ + ly) * 72 + (wx * WSZ + lx);
          ((float*)outp)[(size_t)orig * N + gn] = res[(size_t)orig * N + gn] + v;
        } else {
          ((float*)outp)[(size_t)gm * N + gn] = res[(size_t)gm * N + gn] + v;
        }
      }
    }
  }
}

// ---------------------------------------------------------------------------
// Flash attention per (window, head, 64-query-row block).  4 waves x 16 rows.
// 64 keys per iteration (16 WMMAs between barriers); online softmax; P re-
// staged through per-wave LDS into WMMA A-layout.
// ---------------------------------------------------------------------------
__global__ __launch_bounds__(128)
void attn_kernel(const bf16_t* __restrict__ q, const bf16_t* __restrict__ k,
                 const bf16_t* __restrict__ v, bf16_t* __restrict__ o) {
  __shared__ alignas(64) bf16_t Klds[64][64];       // [key][hd]
  __shared__ alignas(64) bf16_t Vt[64][64];         // transposed: [hd][key]
  __shared__ alignas(64) bf16_t Plds[4][16][64];    // per-wave P staging

  const int tid = threadIdx.x;
  const int lane = tid & 31;
  const int wave = tid >> 5;
  const int blk = blockIdx.x;                       // ((w*NH + h)*9 + qb)
  const int qb = blk % 9;
  const int wh = blk / 9;
  const int h = wh % NH;
  const int w = wh / NH;

  const int qrow0 = w * S_WIN + qb * 64 + wave * 16;
  const int col0 = h * HD;
  const int lm = lane & 15;
  const int lg = lane >> 4;

  // Q fragments for both 32-wide K-dim chunks (HD = 64)
  v16bf aq[2];
  {
    const bf16_t* qrow = q + (size_t)(qrow0 + lm) * HS + col0;
    aq[0] = load_a_frag(qrow, lane);
    aq[1] = load_a_frag(qrow + 32, lane);
  }

  v8f oacc[4];
#pragma unroll
  for (int i = 0; i < 4; i++) oacc[i] = v8f{0, 0, 0, 0, 0, 0, 0, 0};
  float mrow[8], lrow[8];
#pragma unroll
  for (int r = 0; r < 8; r++) { mrow[r] = -1e30f; lrow[r] = 0.f; }

  const int li = tid >> 1;                          // loader row 0..63
  const int ld0 = (tid & 1) * 32;                   // loader col 0 / 32

  for (int kb = 0; kb < S_WIN; kb += 64) {
    __syncthreads();
    {   // cooperative load of K tile (row-major) and V tile (transposed)
      size_t src = (size_t)(w * S_WIN + kb + li) * HS + col0 + ld0;
#pragma unroll
      for (int c = 0; c < 2; c++) {
        *(v16bf*)&Klds[li][ld0 + c * 16] = *(const v16bf*)(k + src + c * 16);
        v16bf vv = *(const v16bf*)(v + src + c * 16);
#pragma unroll
        for (int e = 0; e < 16; e++) Vt[ld0 + c * 16 + e][li] = vv[e];
      }
      if (kb + 64 < S_WIN) {
        __builtin_prefetch(k + src + 64 * HS, 0, 1);
        __builtin_prefetch(v + src + 64 * HS, 0, 1);
      }
    }
    __syncthreads();

    // scores for four 16-key tiles: S = Q(16x64) x K^T
    v8f sc[4];
#pragma unroll
    for (int t = 0; t < 4; t++) sc[t] = v8f{0, 0, 0, 0, 0, 0, 0, 0};
#pragma unroll
    for (int c = 0; c < 2; c++) {
#pragma unroll
      for (int t = 0; t < 4; t++) {
        v16bf bk = *(const v16bf*)&Klds[t * 16 + lm][c * 32 + lg * 16];
        sc[t] = wmma_bf16(aq[c], bk, sc[t]);
      }
    }

    // online softmax update (rows live in VGPR index r + lane-half lg)
#pragma unroll
    for (int r = 0; r < 8; r++) {
      float e[4];
      float mx = -1e30f;
#pragma unroll
      for (int t = 0; t < 4; t++) {
        e[t] = sc[t][r] * 0.125f;                    // HD^-0.5
        mx = fmaxf(mx, e[t]);
      }
#pragma unroll
      for (int off = 1; off < 16; off <<= 1) mx = fmaxf(mx, __shfl_xor(mx, off, 16));
      float mnew = fmaxf(mrow[r], mx);
      float corr = __expf(mrow[r] - mnew);
      mrow[r] = mnew;
      float rs = 0.f;
#pragma unroll
      for (int t = 0; t < 4; t++) {
        e[t] = __expf(e[t] - mnew);
        rs += e[t];
      }
#pragma unroll
      for (int off = 1; off < 16; off <<= 1) rs += __shfl_xor(rs, off, 16);
      lrow[r] = lrow[r] * corr + rs;
#pragma unroll
      for (int ni = 0; ni < 4; ni++) oacc[ni][r] *= corr;
      int m = r + lg * 8;
#pragma unroll
      for (int t = 0; t < 4; t++) Plds[wave][m][t * 16 + lm] = (bf16_t)e[t];
    }

    // O += P(16x64) x V(64x64)
    v16bf ap0 = load_a_frag(&Plds[wave][lm][0], lane);
    v16bf ap1 = load_a_frag(&Plds[wave][lm][32], lane);
#pragma unroll
    for (int ni = 0; ni < 4; ni++) {
      v16bf bv0 = *(const v16bf*)&Vt[ni * 16 + lm][lg * 16];
      v16bf bv1 = *(const v16bf*)&Vt[ni * 16 + lm][32 + lg * 16];
      oacc[ni] = wmma_bf16(ap0, bv0, oacc[ni]);
      oacc[ni] = wmma_bf16(ap1, bv1, oacc[ni]);
    }
  }

  // normalize and store (window-ordered)
#pragma unroll
  for (int ni = 0; ni < 4; ni++) {
#pragma unroll
    for (int r = 0; r < 8; r++) {
      float val = oacc[ni][r] / lrow[r];
      int gm = qrow0 + r + lg * 8;
      o[(size_t)gm * HS + col0 + ni * 16 + lm] = (bf16_t)val;
    }
  }
}

// ---------------------------------------------------------------------------
// Host-side launch
// ---------------------------------------------------------------------------
extern "C" void kernel_launch(void* const* d_in, const int* in_sizes, int n_in,
                              void* d_out, int out_size, void* d_ws, size_t ws_size,
                              hipStream_t stream) {
  const float* hs   = (const float*)d_in[0];
  const float* ln1g = (const float*)d_in[1];
  const float* ln1b = (const float*)d_in[2];
  const float* wq = (const float*)d_in[3];  const float* bq = (const float*)d_in[4];
  const float* wk = (const float*)d_in[5];  const float* bk = (const float*)d_in[6];
  const float* wv = (const float*)d_in[7];  const float* bv = (const float*)d_in[8];
  const float* wo = (const float*)d_in[9];  const float* bo = (const float*)d_in[10];
  const float* ln2g = (const float*)d_in[11]; const float* ln2b = (const float*)d_in[12];
  const float* w1 = (const float*)d_in[13]; const float* b1 = (const float*)d_in[14];
  const float* w2 = (const float*)d_in[15]; const float* b2 = (const float*)d_in[16];
  (void)in_sizes; (void)n_in; (void)out_size; (void)ws_size;

  char* p = (char*)d_ws;
  auto carve = [&](size_t nbytes) -> char* {
    char* r = p;
    p += (nbytes + 255) & ~(size_t)255;
    return r;
  };
  bf16_t* wq_bf = (bf16_t*)carve((size_t)HS * HS * 2);
  bf16_t* wk_bf = (bf16_t*)carve((size_t)HS * HS * 2);
  bf16_t* wv_bf = (bf16_t*)carve((size_t)HS * HS * 2);
  bf16_t* wo_bf = (bf16_t*)carve((size_t)HS * HS * 2);
  bf16_t* w1_bf = (bf16_t*)carve((size_t)HS * FFN * 2);
  bf16_t* w2_bf = (bf16_t*)carve((size_t)FFN * HS * 2);
  bf16_t* x_bf  = (bf16_t*)carve((size_t)NTOK * HS * 2);   // LN1, window order
  bf16_t* q_bf  = (bf16_t*)carve((size_t)NTOK * HS * 2);
  bf16_t* k_bf  = (bf16_t*)carve((size_t)NTOK * HS * 2);
  bf16_t* v_bf  = (bf16_t*)carve((size_t)NTOK * HS * 2);
  bf16_t* o_bf  = (bf16_t*)carve((size_t)NTOK * HS * 2);
  bf16_t* y_bf  = (bf16_t*)carve((size_t)NTOK * HS * 2);   // LN2 out
  bf16_t* h_bf  = (bf16_t*)carve((size_t)NTOK * FFN * 2);  // MLP hidden (GELU'd)
  float*  x_f32 = (float*)carve((size_t)NTOK * HS * 4);    // post-attn residual

  // 1) weight conversions
  auto cvt = [&](const float* s, bf16_t* d, int n) {
    cvt_bf16_kernel<<<(n + 255) / 256, 256, 0, stream>>>(s, d, n);
  };
  cvt(wq, wq_bf, HS * HS);   cvt(wk, wk_bf, HS * HS);
  cvt(wv, wv_bf, HS * HS);   cvt(wo, wo_bf, HS * HS);
  cvt(w1, w1_bf, HS * FFN);  cvt(w2, w2_bf, FFN * HS);

  // 2) LN1 fused with window partition
  ln_kernel<true><<<NTOK, 256, 0, stream>>>(hs, ln1g, ln1b, x_bf);

  // 3) QKV projections (bf16 WMMA GEMMs)
  dim3 gq(HS / 128, NTOK / 128);
  gemm_kernel<0><<<gq, 256, 0, stream>>>(x_bf, wq_bf, bq, q_bf, nullptr, NTOK, HS, HS);
  gemm_kernel<0><<<gq, 256, 0, stream>>>(x_bf, wk_bf, bk, k_bf, nullptr, NTOK, HS, HS);
  gemm_kernel<0><<<gq, 256, 0, stream>>>(x_bf, wv_bf, bv, v_bf, nullptr, NTOK, HS, HS);

  // 4) RoPE on q and k (in place)
  int rope_threads = NTOK * NH * 32;
  rope_kernel<<<(rope_threads + 255) / 256, 256, 0, stream>>>(q_bf);
  rope_kernel<<<(rope_threads + 255) / 256, 256, 0, stream>>>(k_bf);

  // 5) windowed flash attention
  attn_kernel<<<18 * NH * 9, 128, 0, stream>>>(q_bf, k_bf, v_bf, o_bf);

  // 6) output projection + residual + window unpartition -> x_f32
  gemm_kernel<2><<<gq, 256, 0, stream>>>(o_bf, wo_bf, bo, x_f32, hs, NTOK, HS, HS);

  // 7) LN2
  ln_kernel<false><<<NTOK, 256, 0, stream>>>(x_f32, ln2g, ln2b, y_bf);

  // 8) MLP up + exact GELU
  dim3 g1(FFN / 128, NTOK / 128);
  gemm_kernel<1><<<g1, 256, 0, stream>>>(y_bf, w1_bf, b1, h_bf, nullptr, NTOK, FFN, HS);

  // 9) MLP down + residual -> d_out
  gemm_kernel<3><<<gq, 256, 0, stream>>>(h_bf, w2_bf, b2, d_out, x_f32, NTOK, HS, FFN);
}
